// EfficientFrequencyAttention_41412074668291
// MI455X (gfx1250) — compile-verified, matrix operand order checked by reading, not run on hardware
//
#include <hip/hip_runtime.h>
#include <hip/hip_bf16.h>
#include <math.h>

// ---------------------------------------------------------------------------
// EfficientFrequencyAttention for MI455X (gfx1250, wave32, WMMA 16x16x32 f16)
// B=8, C=256, H=W=64, HW=4096
// ---------------------------------------------------------------------------

typedef __attribute__((ext_vector_type(16))) _Float16 v16h;
typedef __attribute__((ext_vector_type(8)))  _Float16 v8h;
typedef __attribute__((ext_vector_type(8)))  float    v8f;

#define BB 8
#define CC 256
#define HH 64
#define WW 64
#define HWN 4096

// ---------------------------------------------------------------------------
// NT GEMM: D[M x N] (f32) = A[M x K] (f16, row-major) * Bt[N x K]^T (f16) + bias
// Block: 32x8 = 8 waves. Each wave: 16 rows x 64 cols (4 WMMA accumulators).
// Block tile: 128 (M) x 64 (N). Fragments per the CDNA5 16-bit A/B layout:
// lanes 0-15 hold K = k0+0..7 and k0+16..23; lanes 16-31 hold K = k0+8..15, 24..31.
// Row pointers are 16B aligned (K multiple of 32, base 256B aligned).
// ---------------------------------------------------------------------------
__global__ __launch_bounds__(256) void gemm_nt_f16(
    const _Float16* __restrict__ A, long long sA,
    const _Float16* __restrict__ Bt, long long sB,
    float* __restrict__ D, long long sD,
    int M, int N, int K, const float* __restrict__ bias)
{
    const int lane = threadIdx.x;   // 0..31
    const int wave = threadIdx.y;   // 0..7
    const int b    = blockIdx.z;

    A  += (size_t)b * (size_t)sA;
    Bt += (size_t)b * (size_t)sB;
    D  += (size_t)b * (size_t)sD;

    const int m0     = blockIdx.y * 128 + wave * 16;
    const int n0     = blockIdx.x * 64;
    const int lanelo = lane & 15;
    const int sel    = (lane < 16) ? 0 : 8;   // which K-octet this lane owns

    v8f acc0 = {}, acc1 = {}, acc2 = {}, acc3 = {};

    const _Float16* __restrict__ arow  = A  + (size_t)(m0 + lanelo) * (size_t)K;
    const _Float16* __restrict__ brow0 = Bt + (size_t)(n0 + lanelo + 0)  * (size_t)K;
    const _Float16* __restrict__ brow1 = Bt + (size_t)(n0 + lanelo + 16) * (size_t)K;
    const _Float16* __restrict__ brow2 = Bt + (size_t)(n0 + lanelo + 32) * (size_t)K;
    const _Float16* __restrict__ brow3 = Bt + (size_t)(n0 + lanelo + 48) * (size_t)K;

    for (int k0 = 0; k0 < K; k0 += 32) {
        v16h af;
        {
            v8h lo = *(const v8h*)(arow + k0 + sel);
            v8h hi = *(const v8h*)(arow + k0 + 16 + sel);
            #pragma unroll
            for (int i = 0; i < 8; ++i) { af[i] = lo[i]; af[i + 8] = hi[i]; }
        }
        v16h bf0, bf1, bf2, bf3;
        {
            v8h lo = *(const v8h*)(brow0 + k0 + sel);
            v8h hi = *(const v8h*)(brow0 + k0 + 16 + sel);
            #pragma unroll
            for (int i = 0; i < 8; ++i) { bf0[i] = lo[i]; bf0[i + 8] = hi[i]; }
        }
        {
            v8h lo = *(const v8h*)(brow1 + k0 + sel);
            v8h hi = *(const v8h*)(brow1 + k0 + 16 + sel);
            #pragma unroll
            for (int i = 0; i < 8; ++i) { bf1[i] = lo[i]; bf1[i + 8] = hi[i]; }
        }
        {
            v8h lo = *(const v8h*)(brow2 + k0 + sel);
            v8h hi = *(const v8h*)(brow2 + k0 + 16 + sel);
            #pragma unroll
            for (int i = 0; i < 8; ++i) { bf2[i] = lo[i]; bf2[i + 8] = hi[i]; }
        }
        {
            v8h lo = *(const v8h*)(brow3 + k0 + sel);
            v8h hi = *(const v8h*)(brow3 + k0 + 16 + sel);
            #pragma unroll
            for (int i = 0; i < 8; ++i) { bf3[i] = lo[i]; bf3[i + 8] = hi[i]; }
        }
        acc0 = __builtin_amdgcn_wmma_f32_16x16x32_f16(false, af, false, bf0, (short)0, acc0, false, false);
        acc1 = __builtin_amdgcn_wmma_f32_16x16x32_f16(false, af, false, bf1, (short)0, acc1, false, false);
        acc2 = __builtin_amdgcn_wmma_f32_16x16x32_f16(false, af, false, bf2, (short)0, acc2, false, false);
        acc3 = __builtin_amdgcn_wmma_f32_16x16x32_f16(false, af, false, bf3, (short)0, acc3, false, false);
    }

    // C/D layout: lanes 0-15 -> (M = m0+r,   N = n0+lane); lanes 16-31 -> (M = m0+8+r)
    const int mbase = m0 + ((lane < 16) ? 0 : 8);
    const int ncol  = n0 + lanelo;
    #pragma unroll
    for (int r = 0; r < 8; ++r) {
        const int m = mbase + r;
        const float bv = bias ? bias[m] : 0.0f;
        D[(size_t)m * (size_t)N + (size_t)(ncol + 0)]  = acc0[r] + bv;
        D[(size_t)m * (size_t)N + (size_t)(ncol + 16)] = acc1[r] + bv;
        D[(size_t)m * (size_t)N + (size_t)(ncol + 32)] = acc2[r] + bv;
        D[(size_t)m * (size_t)N + (size_t)(ncol + 48)] = acc3[r] + bv;
    }
}

// ---------------------------------------------------------------------------
// Tiled transpose with f32 -> f16 conversion: in[R x Cc] -> out[Cc x R]
// ---------------------------------------------------------------------------
__global__ __launch_bounds__(256) void transpose_f32_to_f16(
    const float* __restrict__ in, long long sIn,
    _Float16* __restrict__ out, long long sOut, int R, int Cc)
{
    __shared__ float tile[32][33];
    const int b = blockIdx.z;
    in  += (size_t)b * (size_t)sIn;
    out += (size_t)b * (size_t)sOut;
    const int c0 = blockIdx.x * 32;
    const int r0 = blockIdx.y * 32;
    const int tx = threadIdx.x;   // 0..31
    const int ty = threadIdx.y;   // 0..7
    #pragma unroll
    for (int i = 0; i < 32; i += 8)
        tile[ty + i][tx] = in[(size_t)(r0 + ty + i) * (size_t)Cc + (size_t)(c0 + tx)];
    __syncthreads();
    #pragma unroll
    for (int i = 0; i < 32; i += 8)
        out[(size_t)(c0 + ty + i) * (size_t)R + (size_t)(r0 + tx)] = (_Float16)tile[tx][ty + i];
}

// ---------------------------------------------------------------------------
// Elementwise f32 -> f16 convert
// ---------------------------------------------------------------------------
__global__ __launch_bounds__(256) void cvt_f32_f16(
    const float* __restrict__ in, _Float16* __restrict__ out, size_t n)
{
    size_t i = (size_t)blockIdx.x * 256 + threadIdx.x;
    if (i < n) out[i] = (_Float16)in[i];
}

// ---------------------------------------------------------------------------
// Row softmax (row length L, one block per row), f32 in -> f16 out
// ---------------------------------------------------------------------------
__global__ __launch_bounds__(256) void row_softmax_f32_to_f16(
    const float* __restrict__ in, _Float16* __restrict__ out, int L)
{
    __shared__ float red[256];
    const size_t row = blockIdx.x;
    in  += row * (size_t)L;
    out += row * (size_t)L;
    const int t = threadIdx.x;

    float mx = -INFINITY;
    for (int i = t; i < L; i += 256) mx = fmaxf(mx, in[i]);
    red[t] = mx; __syncthreads();
    for (int s = 128; s > 0; s >>= 1) { if (t < s) red[t] = fmaxf(red[t], red[t + s]); __syncthreads(); }
    mx = red[0]; __syncthreads();

    float sum = 0.0f;
    for (int i = t; i < L; i += 256) sum += __expf(in[i] - mx);
    red[t] = sum; __syncthreads();
    for (int s = 128; s > 0; s >>= 1) { if (t < s) red[t] += red[t + s]; __syncthreads(); }
    const float inv = 1.0f / red[0];

    for (int i = t; i < L; i += 256) out[i] = (_Float16)(__expf(in[i] - mx) * inv);
}

// ---------------------------------------------------------------------------
// Column softmax over C (axis=1 of [B, C, HW]), in-place f32 (queries)
// ---------------------------------------------------------------------------
__global__ __launch_bounds__(256) void col_softmax_inplace(
    float* __restrict__ x, int C, int HW)
{
    const size_t b = blockIdx.y;
    const int n = blockIdx.x * 256 + threadIdx.x;
    if (n >= HW) return;
    float* p = x + b * (size_t)C * (size_t)HW + (size_t)n;
    float mx = -INFINITY;
    for (int c = 0; c < C; ++c) mx = fmaxf(mx, p[(size_t)c * HW]);
    float sum = 0.0f;
    for (int c = 0; c < C; ++c) sum += __expf(p[(size_t)c * HW] - mx);
    const float inv = 1.0f / sum;
    for (int c = 0; c < C; ++c) p[(size_t)c * HW] = __expf(p[(size_t)c * HW] - mx) * inv;
}

// ---------------------------------------------------------------------------
// Row softmax over 256-wide rows of [B*256, 256] with accumulate into `accum`
// (the per-level _chan_att softmax(A, axis=-1), summed across pyramid levels)
// ---------------------------------------------------------------------------
__global__ __launch_bounds__(256) void row_softmax256_accum(
    const float* __restrict__ in, float* __restrict__ accum, int init)
{
    __shared__ float red[256];
    const size_t row = blockIdx.x;
    const int t = threadIdx.x;
    const float v = in[row * 256 + t];
    red[t] = v; __syncthreads();
    for (int s = 128; s > 0; s >>= 1) { if (t < s) red[t] = fmaxf(red[t], red[t + s]); __syncthreads(); }
    const float mx = red[0]; __syncthreads();
    const float e = __expf(v - mx);
    red[t] = e; __syncthreads();
    for (int s = 128; s > 0; s >>= 1) { if (t < s) red[t] += red[t + s]; __syncthreads(); }
    const float r = e / red[0];
    accum[row * 256 + t] = init ? r : (accum[row * 256 + t] + r);
}

// ---------------------------------------------------------------------------
// Depthwise Gaussian conv ('SAME', zero pad). Optionally also emit Laplacian
// lap = in - conv(in). Weights passed by value (computed on host).
// ---------------------------------------------------------------------------
struct GaussK { float w[25]; };

__global__ __launch_bounds__(256) void dw_gauss(
    const float* __restrict__ in, float* __restrict__ gout,
    float* __restrict__ lap, int H, int W, int k, GaussK gk)
{
    const size_t plane = blockIdx.y;           // B*C planes
    const int idx = blockIdx.x * 256 + threadIdx.x;
    if (idx >= H * W) return;
    const int y = idx / W, x = idx % W;
    const float* __restrict__ src = in + plane * (size_t)(H * W);
    const int r = k >> 1;
    float s = 0.0f;
    for (int dy = 0; dy < k; ++dy) {
        const int yy = y + dy - r;
        if (yy < 0 || yy >= H) continue;
        for (int dx = 0; dx < k; ++dx) {
            const int xx = x + dx - r;
            if (xx < 0 || xx >= W) continue;
            s += gk.w[dy * k + dx] * src[yy * W + xx];
        }
    }
    if (gout) gout[plane * (size_t)(H * W) + idx] = s;
    if (lap)  lap [plane * (size_t)(H * W) + idx] = src[idx] - s;
}

// ---------------------------------------------------------------------------
// Final depthwise Conv3d(2,1,1): out = w_dw[c,0]*eff + w_dw[c,1]*freq
// ---------------------------------------------------------------------------
__global__ __launch_bounds__(256) void combine_out(
    const float* __restrict__ eff, const float* __restrict__ freq,
    const float* __restrict__ w_dw, float* __restrict__ out, size_t total, int C, int HW)
{
    size_t i = (size_t)blockIdx.x * 256 + threadIdx.x;
    if (i >= total) return;
    const int c = (int)((i / (size_t)HW) % (size_t)C);
    out[i] = w_dw[c * 2 + 0] * eff[i] + w_dw[c * 2 + 1] * freq[i];
}

// ---------------------------------------------------------------------------
static void gauss2d_host(int k, double sigma, GaussK* gk)
{
    double g[8]; double s = 0.0;
    for (int i = 0; i < k; ++i) {
        const double d = (double)i - (k - 1) / 2.0;
        g[i] = exp(-(d * d) / (2.0 * sigma * sigma));
        s += g[i];
    }
    for (int i = 0; i < k; ++i) g[i] /= s;
    for (int i = 0; i < k; ++i)
        for (int j = 0; j < k; ++j)
            gk->w[i * k + j] = (float)(g[i] * g[j]);
}

extern "C" void kernel_launch(void* const* d_in, const int* in_sizes, int n_in,
                              void* d_out, int out_size, void* d_ws, size_t ws_size,
                              hipStream_t stream)
{
    (void)in_sizes; (void)n_in; (void)out_size; (void)ws_size;

    const float* x   = (const float*)d_in[0];
    const float* Wk  = (const float*)d_in[1];
    const float* bk  = (const float*)d_in[2];
    const float* Wq  = (const float*)d_in[3];
    const float* bq  = (const float*)d_in[4];
    const float* Wv  = (const float*)d_in[5];
    const float* bv  = (const float*)d_in[6];
    const float* Wr  = (const float*)d_in[7];
    const float* br  = (const float*)d_in[8];
    const float* wdw = (const float*)d_in[9];
    float* out = (float*)d_out;

    const long long PLANE = (long long)BB * CC * HWN;          // f32 elements per full tensor
    const long long CHW   = (long long)CC * HWN;               // per-batch plane
    const long long SQ    = (long long)CC * CC;                // 256x256

    // Workspace layout (bytes; all naturally aligned, sizes are power-of-2-ish)
    char* ws = (char*)d_ws;
    size_t off = 0;
    float*    F0    = (float*)(ws + off);    off += (size_t)PLANE * 4;   // scratch plane
    float*    F1    = (float*)(ws + off);    off += (size_t)PLANE * 4;   // eff
    float*    F2    = (float*)(ws + off);    off += (size_t)PLANE * 4;   // laplacian scratch
    _Float16* Hx16  = (_Float16*)(ws + off); off += (size_t)PLANE * 2;   // x f16 [B,C,HW]
    _Float16* HxT16 = (_Float16*)(ws + off); off += (size_t)PLANE * 2;   // x^T f16 [B,HW,C] -> reused as attended^T
    _Float16* Hk16  = (_Float16*)(ws + off); off += (size_t)PLANE * 2;   // softmaxed keys f16
    _Float16* HqT16 = (_Float16*)(ws + off); off += (size_t)PLANE * 2;   // queries^T f16
    _Float16* Hv16  = (_Float16*)(ws + off); off += (size_t)PLANE * 2;   // values f16
    _Float16* HL16  = (_Float16*)(ws + off); off += (size_t)PLANE * 2;   // laplacian f16
    _Float16* HLs16 = (_Float16*)(ws + off); off += (size_t)PLANE * 2;   // softmaxed laplacian f16
    float*    Sctx  = (float*)(ws + off);    off += (size_t)BB * SQ * 4; // context / gram scratch
    float*    Satt  = (float*)(ws + off);    off += (size_t)BB * SQ * 4; // accumulated channel att
    _Float16* ShT   = (_Float16*)(ws + off); off += (size_t)BB * SQ * 2; // ctx^T / att^T f16
    _Float16* Wk16  = (_Float16*)(ws + off); off += (size_t)SQ * 2;
    _Float16* Wq16  = (_Float16*)(ws + off); off += (size_t)SQ * 2;
    _Float16* Wv16  = (_Float16*)(ws + off); off += (size_t)SQ * 2;
    _Float16* Wr16  = (_Float16*)(ws + off); off += (size_t)SQ * 2;

    const dim3 blkG(32, 8);                       // 8 waves
    const dim3 gemmWide((HWN) / 64, CC / 128, BB);  // N=4096 GEMMs
    const dim3 gemmSq(CC / 64, CC / 128, BB);       // N=256 GEMMs (context/gram)
    const dim3 trX(HWN / 32, CC / 32, BB);          // [256 x 4096] -> [4096 x 256]
    const dim3 trSq(CC / 32, CC / 32, BB);          // [256 x 256] transpose
    const int  cvtBlocks = (int)((PLANE + 255) / 256);

    // --- Weight conversion to f16 -------------------------------------------
    cvt_f32_f16<<<(SQ + 255) / 256, 256, 0, stream>>>(Wk, Wk16, (size_t)SQ);
    cvt_f32_f16<<<(SQ + 255) / 256, 256, 0, stream>>>(Wq, Wq16, (size_t)SQ);
    cvt_f32_f16<<<(SQ + 255) / 256, 256, 0, stream>>>(Wv, Wv16, (size_t)SQ);
    cvt_f32_f16<<<(SQ + 255) / 256, 256, 0, stream>>>(Wr, Wr16, (size_t)SQ);

    // --- x -> f16 (plain + transposed) --------------------------------------
    cvt_f32_f16<<<cvtBlocks, 256, 0, stream>>>(x, Hx16, (size_t)PLANE);
    transpose_f32_to_f16<<<trX, blkG, 0, stream>>>(x, CHW, HxT16, CHW, CC, HWN);

    // --- values = Wv x + bv --------------------------------------------------
    gemm_nt_f16<<<gemmWide, blkG, 0, stream>>>(Wv16, 0, HxT16, CHW, F0, CHW, CC, HWN, CC, bv);
    cvt_f32_f16<<<cvtBlocks, 256, 0, stream>>>(F0, Hv16, (size_t)PLANE);

    // --- keys = softmax_HW(Wk x + bk) ---------------------------------------
    gemm_nt_f16<<<gemmWide, blkG, 0, stream>>>(Wk16, 0, HxT16, CHW, F0, CHW, CC, HWN, CC, bk);
    row_softmax_f32_to_f16<<<BB * CC, 256, 0, stream>>>(F0, Hk16, HWN);

    // --- queries = softmax_C(Wq x + bq), then transpose ----------------------
    gemm_nt_f16<<<gemmWide, blkG, 0, stream>>>(Wq16, 0, HxT16, CHW, F0, CHW, CC, HWN, CC, bq);
    col_softmax_inplace<<<dim3(HWN / 256, BB), 256, 0, stream>>>(F0, CC, HWN);
    transpose_f32_to_f16<<<trX, blkG, 0, stream>>>(F0, CHW, HqT16, CHW, CC, HWN);

    // --- context[k,v] = keys . values^T  (K = HW = 4096) ---------------------
    gemm_nt_f16<<<gemmSq, blkG, 0, stream>>>(Hk16, CHW, Hv16, CHW, Sctx, SQ, CC, CC, HWN, nullptr);
    transpose_f32_to_f16<<<trSq, blkG, 0, stream>>>(Sctx, SQ, ShT, SQ, CC, CC);

    // --- attended = context^T . queries --------------------------------------
    gemm_nt_f16<<<gemmWide, blkG, 0, stream>>>(ShT, SQ, HqT16, CHW, F0, CHW, CC, HWN, CC, nullptr);
    transpose_f32_to_f16<<<trX, blkG, 0, stream>>>(F0, CHW, HxT16, CHW, CC, HWN);  // reuse xT as attended^T

    // --- eff = Wr attended + br ----------------------------------------------
    gemm_nt_f16<<<gemmWide, blkG, 0, stream>>>(Wr16, 0, HxT16, CHW, F1, CHW, CC, HWN, CC, br);

    // --- channel attention, level 0 (on x itself) ----------------------------
    row_softmax_f32_to_f16<<<BB * CC, 256, 0, stream>>>(x, HLs16, HWN);
    gemm_nt_f16<<<gemmSq, blkG, 0, stream>>>(HLs16, CHW, Hx16, CHW, Sctx, SQ, CC, CC, HWN, nullptr);
    row_softmax256_accum<<<BB * CC, 256, 0, stream>>>(Sctx, Satt, 1);

    // --- Laplacian pyramid levels -------------------------------------------
    GaussK g3, g5;
    gauss2d_host(3, 1.6, &g3);
    gauss2d_host(5, 1.6 * pow(2.0, 1.0 / 3.0), &g5);
    const dim3 convGrid((HH * WW) / 256, BB * CC);

    // level 1: G1 = gauss3(x); L1 = x - G1
    dw_gauss<<<convGrid, 256, 0, stream>>>(x, F0, F2, HH, WW, 3, g3);
    row_softmax_f32_to_f16<<<BB * CC, 256, 0, stream>>>(F2, HLs16, HWN);
    cvt_f32_f16<<<cvtBlocks, 256, 0, stream>>>(F2, HL16, (size_t)PLANE);
    gemm_nt_f16<<<gemmSq, blkG, 0, stream>>>(HLs16, CHW, HL16, CHW, Sctx, SQ, CC, CC, HWN, nullptr);
    row_softmax256_accum<<<BB * CC, 256, 0, stream>>>(Sctx, Satt, 0);

    // level 2: G2 = gauss5(G1); L2 = G1 - G2 (G2 discarded)
    dw_gauss<<<convGrid, 256, 0, stream>>>(F0, nullptr, F2, HH, WW, 5, g5);
    row_softmax_f32_to_f16<<<BB * CC, 256, 0, stream>>>(F2, HLs16, HWN);
    cvt_f32_f16<<<cvtBlocks, 256, 0, stream>>>(F2, HL16, (size_t)PLANE);
    gemm_nt_f16<<<gemmSq, blkG, 0, stream>>>(HLs16, CHW, HL16, CHW, Sctx, SQ, CC, CC, HWN, nullptr);
    row_softmax256_accum<<<BB * CC, 256, 0, stream>>>(Sctx, Satt, 0);

    // --- freq = att^T . queries ----------------------------------------------
    transpose_f32_to_f16<<<trSq, blkG, 0, stream>>>(Satt, SQ, ShT, SQ, CC, CC);
    gemm_nt_f16<<<gemmWide, blkG, 0, stream>>>(ShT, SQ, HqT16, CHW, F0, CHW, CC, HWN, CC, nullptr);

    // --- out = w_dw[:,0]*eff + w_dw[:,1]*freq --------------------------------
    combine_out<<<cvtBlocks, 256, 0, stream>>>(F1, F0, wdw, out, (size_t)PLANE, CC, HWN);
}
